// LHMSA_62723702391244
// MI455X (gfx1250) — compile-verified, matrix operand order checked by reading, not run on hardware
//
#include <hip/hip_runtime.h>
#include <hip/hip_bf16.h>

typedef __attribute__((ext_vector_type(16))) _Float16 v16h;
typedef __attribute__((ext_vector_type(8)))  float    v8f;

#define WMMA_F16(a, b, c) \
  __builtin_amdgcn_wmma_f32_16x16x32_f16(false, (a), false, (b), (short)0, (c), false, false)

#if __has_builtin(__builtin_amdgcn_global_load_async_to_lds_b32) && \
    __has_builtin(__builtin_amdgcn_s_wait_asynccnt)
#define HAVE_ASYNC_LDS 1
#else
#define HAVE_ASYNC_LDS 0
#endif

// ---------------------------------------------------------------------------
// Fragment loader: A-matrix 16x32 f16 layout (ISA 7.12.2). Also loads B
// fragments from a B^T row-major (N x K) array (identical per-lane pattern).
// ---------------------------------------------------------------------------
__device__ __forceinline__ v16h ldfrag(const _Float16* base, int row0, int ld,
                                       int col0, int lane) {
  int r  = row0 + (lane & 15);
  int h8 = (lane >> 4) << 3;
  const _Float16* p = base + r * ld + col0 + h8;
  v16h f;
#pragma unroll
  for (int e = 0; e < 8; ++e) f[e] = p[e];
#pragma unroll
  for (int e = 0; e < 8; ++e) f[8 + e] = p[16 + e];
  return f;
}

// ---------------------------------------------------------------------------
// K1: DWT  x(2,64,512,512) -> ll(2,64,256,256), high(2,192,256,256)
// ---------------------------------------------------------------------------
__global__ __launch_bounds__(256) void dwt_kernel(const float* __restrict__ x,
                                                  float* __restrict__ ll,
                                                  float* __restrict__ high) {
  int id = blockIdx.x * 256 + threadIdx.x;          // 8388608 total
  int w = id & 255, h = (id >> 8) & 255;
  int c = (id >> 16) & 63, b = id >> 22;
  size_t xb = (((size_t)(b * 64 + c) * 512 + 2 * h) * 512) + 2 * w;
  float x1 = x[xb] * 0.5f;
  float x3 = x[xb + 1] * 0.5f;
  float x2 = x[xb + 512] * 0.5f;
  float x4 = x[xb + 513] * 0.5f;
  ll[id] = x1 + x2 + x3 + x4;
  size_t hb = ((size_t)(b * 192 + c) << 16) + (h << 8) + w;
  high[hb]                         = -x1 - x2 + x3 + x4;
  high[hb + ((size_t)64 << 16)]    = -x1 + x2 - x3 + x4;
  high[hb + ((size_t)128 << 16)]   =  x1 - x2 - x3 + x4;
}

// ---------------------------------------------------------------------------
// K2: weight conversion f32 -> f16 (qkv_w 576x192, proj_w 192x192)
// ---------------------------------------------------------------------------
__global__ void cvt_kernel(const float* __restrict__ qkvw,
                           const float* __restrict__ projw,
                           _Float16* __restrict__ qh, _Float16* __restrict__ ph) {
  int i = blockIdx.x * 256 + threadIdx.x;
  if (i < 110592) qh[i] = (_Float16)qkvw[i];
  if (i < 36864)  ph[i] = (_Float16)projw[i];
}

// ---------------------------------------------------------------------------
// K3: GAP over ll per (b,c)
// ---------------------------------------------------------------------------
__global__ __launch_bounds__(256) void gap_kernel(const float* __restrict__ ll,
                                                  float* __restrict__ gap) {
  __shared__ float red[256];
  int bc = blockIdx.x;  // 0..127
  const float* p = ll + ((size_t)bc << 16);
  float s = 0.f;
  for (int i = threadIdx.x; i < 65536; i += 256) s += p[i];
  red[threadIdx.x] = s;
  __syncthreads();
  for (int st = 128; st > 0; st >>= 1) {
    if (threadIdx.x < st) red[threadIdx.x] += red[threadIdx.x + st];
    __syncthreads();
  }
  if (threadIdx.x == 0) gap[bc] = red[0] * (1.f / 65536.f);
}

// ---------------------------------------------------------------------------
// K4: cross gate  cross[b][0..191] = sigmoid(cg2 @ gelu(cg1 @ gap))
// ---------------------------------------------------------------------------
__global__ void cross_kernel(const float* __restrict__ gap,
                             const float* __restrict__ cg1,
                             const float* __restrict__ cg2,
                             float* __restrict__ cross) {
  __shared__ float hbuf[48];
  int b = blockIdx.x, t = threadIdx.x;  // 192 threads
  if (t < 48) {
    float a = 0.f;
    for (int c = 0; c < 64; ++c) a += cg1[t * 64 + c] * gap[b * 64 + c];
    hbuf[t] = 0.5f * a * (1.f + erff(a * 0.70710678118654752f));
  }
  __syncthreads();
  float a = 0.f;
  for (int k = 0; k < 48; ++k) a += cg2[t * 48 + k] * hbuf[k];
  cross[b * 192 + t] = 1.f / (1.f + expf(-a));
}

// ---------------------------------------------------------------------------
// K5: aux branch: refined_ll = ll + gamma_aux * (gelu(dwconv3(ll)) @ pw^T)
// 256 threads = 4 pixels x 64 channels
// ---------------------------------------------------------------------------
__global__ __launch_bounds__(256) void aux_kernel(const float* __restrict__ ll,
                                                  const float* __restrict__ dw,
                                                  const float* __restrict__ pw,
                                                  const float* __restrict__ gA,
                                                  float* __restrict__ rll) {
  __shared__ float tv[4][64];
  int p = threadIdx.x >> 6, c = threadIdx.x & 63;
  int pix = blockIdx.x * 4 + p;
  int b = pix >> 16, h = (pix >> 8) & 255, w = pix & 255;
  const float* base = ll + (((size_t)(b * 64 + c)) << 16);
  float acc = 0.f;
#pragma unroll
  for (int ky = 0; ky < 3; ++ky) {
    int hh = h + ky - 1;
    if (hh < 0 || hh > 255) continue;
#pragma unroll
    for (int kx = 0; kx < 3; ++kx) {
      int ww = w + kx - 1;
      if (ww < 0 || ww > 255) continue;
      acc += base[(hh << 8) + ww] * dw[c * 9 + ky * 3 + kx];
    }
  }
  float center = base[(h << 8) + w];
  tv[p][c] = 0.5f * acc * (1.f + erff(acc * 0.70710678118654752f));
  __syncthreads();
  float o = 0.f;
  for (int c2 = 0; c2 < 64; ++c2) o += pw[c * 64 + c2] * tv[p][c2];
  rll[(((size_t)(b * 64 + c)) << 16) + (h << 8) + w] = center + gA[0] * o;
}

// ---------------------------------------------------------------------------
// K6: fused LN + shifted-window MHSA + proj + residual (in-place on high).
// One block = one window (64 tokens x 192 ch). 128 threads = 4 waves,
// wave wv owns m-tile wv. All GEMMs via v_wmma_f32_16x16x32_f16.
// Window staging uses GLOBAL_LOAD_ASYNC_TO_LDS (ASYNCcnt) when available.
// ---------------------------------------------------------------------------
__global__ __launch_bounds__(128) void attn_kernel(
    float* __restrict__ high,                  // refined in place
    const float* __restrict__ norm_w, const float* __restrict__ norm_b,
    const _Float16* __restrict__ qkvw,         // (576,192) f16 = B^T row-major
    const float* __restrict__ qkv_b,
    const _Float16* __restrict__ projw,        // (192,192) f16
    const float* __restrict__ proj_b,
    const float* __restrict__ rpb,             // (225,6)
    const int* __restrict__ rpi,               // (64,64)
    const float* __restrict__ gamma_main) {
  __shared__ __align__(16) unsigned char smem[86528];
  _Float16* xs    = (_Float16*)(smem);             // [64][192] normalized f16
  _Float16* aouts = (_Float16*)(smem + 24576);     // [64][192] head outputs
  _Float16* qs    = (_Float16*)(smem + 49152);     // [64][32]
  _Float16* ksm   = (_Float16*)(smem + 53248);     // [64][32]
  _Float16* vts   = (_Float16*)(smem + 57344);     // [32][64]  (v transposed)
  float*    Ss    = (float*)   (smem + 61440);     // [64][64] scores f32
  _Float16* ah    = (_Float16*)(smem + 77824);     // [64][64] attn f16
  float*    stats = (float*)   (smem + 86016);     // [64][2] mean,rstd
  float*    raw   = (float*)   (smem + 24576);     // [64][192] f32 staging (overlay)
  float*    stage = (float*)   (smem + 61440);     // [64][96] out staging (overlay)

  const int tid = threadIdx.x;
  const int lane = tid & 31;
  const int wv = tid >> 5;          // m-tile
  const int m0 = wv * 16;

  const int b = blockIdx.x >> 10;
  const int win = blockIdx.x & 1023;
  const int wy = win >> 5, wx = win & 31;
  const size_t baseB = (size_t)b * 192 * 65536;
  const float gmain = gamma_main[0];

  if (tid == 0) {
    __builtin_prefetch(qkvw, 0, 1);
    __builtin_prefetch(projw, 0, 1);
  }

  // Phase A: stage raw f32 window into LDS (rolled source pixels)
  for (int idx = tid; idx < 64 * 192; idx += 128) {
    int c = idx >> 6;       // channel
    int t = idx & 63;       // token
    int h = (wy * 8 + (t >> 3) + 4) & 255;
    int w = (wx * 8 + (t & 7) + 4) & 255;
    const float* gp = high + baseB + ((size_t)c << 16) + (h << 8) + w;
    float* lp = &raw[t * 192 + c];
#if HAVE_ASYNC_LDS
    __builtin_amdgcn_global_load_async_to_lds_b32(
        (__attribute__((address_space(1))) int*)gp,
        (__attribute__((address_space(3))) int*)lp, 0, 0);
#else
    *lp = *gp;
#endif
  }
#if HAVE_ASYNC_LDS
  __builtin_amdgcn_s_wait_asynccnt(0);
#endif
  __syncthreads();
  // Phase B: LN stats per token
  if (tid < 64) {
    float s = 0.f, s2 = 0.f;
    for (int c = 0; c < 192; ++c) { float v = raw[tid * 192 + c]; s += v; s2 += v * v; }
    float mu = s * (1.f / 192.f);
    float var = s2 * (1.f / 192.f) - mu * mu;
    stats[tid * 2] = mu;
    stats[tid * 2 + 1] = rsqrtf(var + 1e-6f);
  }
  __syncthreads();
  // Phase C: normalize to f16
  for (int idx = tid; idx < 64 * 192; idx += 128) {
    int t = idx / 192, c = idx % 192;
    float v = (raw[t * 192 + c] - stats[t * 2]) * stats[t * 2 + 1] * norm_w[c] + norm_b[c];
    xs[t * 192 + c] = (_Float16)v;
  }
  __syncthreads();

  const int n_lo = lane & 15;
  const int mbase = m0 + ((lane >> 4) << 3);
  const float scale = 0.17677669529663687f;   // 1/sqrt(32)

  for (int hd = 0; hd < 6; ++hd) {
    // ---- q,k,v projections (each: M=64,K=192,N=32) ----
    for (int tgt = 0; tgt < 3; ++tgt) {
      const _Float16* wbase = qkvw + (size_t)(tgt * 192 + hd * 32) * 192;
      v8f acc0 = {}, acc1 = {};
#pragma unroll
      for (int kk = 0; kk < 6; ++kk) {
        v16h a  = ldfrag(xs, m0, 192, kk * 32, lane);
        v16h b0 = ldfrag(wbase, 0, 192, kk * 32, lane);
        v16h b1 = ldfrag(wbase, 16, 192, kk * 32, lane);
        acc0 = WMMA_F16(a, b0, acc0);
        acc1 = WMMA_F16(a, b1, acc1);
      }
      float bz0 = qkv_b[tgt * 192 + hd * 32 + n_lo];
      float bz1 = qkv_b[tgt * 192 + hd * 32 + 16 + n_lo];
#pragma unroll
      for (int r = 0; r < 8; ++r) {
        int m = mbase + r;
        float v0 = acc0[r] + bz0, v1 = acc1[r] + bz1;
        if (tgt == 0)      { qs[m * 32 + n_lo] = (_Float16)v0;  qs[m * 32 + 16 + n_lo] = (_Float16)v1; }
        else if (tgt == 1) { ksm[m * 32 + n_lo] = (_Float16)v0; ksm[m * 32 + 16 + n_lo] = (_Float16)v1; }
        else               { vts[n_lo * 64 + m] = (_Float16)v0; vts[(16 + n_lo) * 64 + m] = (_Float16)v1; }
      }
    }
    __syncthreads();
    // ---- scores S = q @ k^T  (K = dh = 32, one WMMA per tile) ----
    {
      v16h a = ldfrag(qs, m0, 32, 0, lane);
#pragma unroll
      for (int nt = 0; nt < 4; ++nt) {
        v16h bf = ldfrag(ksm, nt * 16, 32, 0, lane);
        v8f s = {};
        s = WMMA_F16(a, bf, s);
        int n = nt * 16 + n_lo;
#pragma unroll
        for (int r = 0; r < 8; ++r) {
          int m = mbase + r;
          int idx = rpi[m * 64 + n];
          Ss[m * 64 + n] = s[r] * scale + rpb[idx * 6 + hd];
        }
      }
    }
    __syncthreads();
    // ---- softmax (one row per thread) ----
    if (tid < 64) {
      float mx = -1e30f;
      for (int n = 0; n < 64; ++n) mx = fmaxf(mx, Ss[tid * 64 + n]);
      float sum = 0.f;
      for (int n = 0; n < 64; ++n) {
        float e = expf(Ss[tid * 64 + n] - mx);
        Ss[tid * 64 + n] = e;
        sum += e;
      }
      float inv = 1.f / sum;
      for (int n = 0; n < 64; ++n) ah[tid * 64 + n] = (_Float16)(Ss[tid * 64 + n] * inv);
    }
    __syncthreads();
    // ---- attn @ v  (M=64,K=64,N=32); B^T = vts (dh x tokens) ----
    {
      v8f acc0 = {}, acc1 = {};
#pragma unroll
      for (int kk = 0; kk < 2; ++kk) {
        v16h a  = ldfrag(ah, m0, 64, kk * 32, lane);
        v16h b0 = ldfrag(vts, 0, 64, kk * 32, lane);
        v16h b1 = ldfrag(vts, 16, 64, kk * 32, lane);
        acc0 = WMMA_F16(a, b0, acc0);
        acc1 = WMMA_F16(a, b1, acc1);
      }
#pragma unroll
      for (int r = 0; r < 8; ++r) {
        int m = mbase + r;
        aouts[m * 192 + hd * 32 + n_lo]      = (_Float16)acc0[r];
        aouts[m * 192 + hd * 32 + 16 + n_lo] = (_Float16)acc1[r];
      }
    }
    __syncthreads();
  }

  // ---- proj (M=64,K=192,N=192) + residual, staged through LDS for
  //      coalesced global RMW (two 96-channel halves) ----
  for (int half = 0; half < 2; ++half) {
    for (int nt = half * 6; nt < half * 6 + 6; ++nt) {
      v8f acc = {};
#pragma unroll
      for (int kk = 0; kk < 6; ++kk) {
        v16h a  = ldfrag(aouts, m0, 192, kk * 32, lane);
        v16h bf = ldfrag(projw, nt * 16, 192, kk * 32, lane);
        acc = WMMA_F16(a, bf, acc);
      }
      int c = nt * 16 + n_lo;
      int cl = c - half * 96;                   // 0..95
      float pb = proj_b[c];
#pragma unroll
      for (int r = 0; r < 8; ++r) {
        int m = mbase + r;
        stage[m * 96 + cl] = acc[r] + pb;
      }
    }
    __syncthreads();
    for (int idx = tid; idx < 64 * 96; idx += 128) {
      int cl = idx >> 6;      // 0..95
      int t = idx & 63;       // token
      int c = half * 96 + cl;
      int h = (wy * 8 + (t >> 3) + 4) & 255;
      int w = (wx * 8 + (t & 7) + 4) & 255;
      size_t gi = baseB + ((size_t)c << 16) + (h << 8) + w;
      high[gi] = high[gi] + gmain * stage[t * 96 + cl];
    }
    __syncthreads();
  }
}

// ---------------------------------------------------------------------------
// K7: IWT + cross gate + final 1x1 conv on (out - x)
// 256 threads = 4 quarter-pixels x 64 channels
// ---------------------------------------------------------------------------
__global__ __launch_bounds__(256) void iwt_kernel(
    const float* __restrict__ rll, const float* __restrict__ rhigh,
    const float* __restrict__ cross, const float* __restrict__ x,
    const float* __restrict__ po, const float* __restrict__ gC,
    float* __restrict__ out) {
  __shared__ float df[4][4][64];
  int p = threadIdx.x >> 6, c = threadIdx.x & 63;
  int qp = blockIdx.x * 4 + p;
  int b = qp >> 16, h = (qp >> 8) & 255, w = qp & 255;
  float llv = rll[(((size_t)(b * 64 + c)) << 16) + (h << 8) + w];
  size_t hb = (((size_t)b * 192) << 16) + (h << 8) + w;
  float g = gC[0];
  float f0 = 1.f + g * (cross[b * 192 + c] - 0.5f) * 2.f;
  float f1 = 1.f + g * (cross[b * 192 + 64 + c] - 0.5f) * 2.f;
  float f2 = 1.f + g * (cross[b * 192 + 128 + c] - 0.5f) * 2.f;
  float hl  = rhigh[hb + ((size_t)c << 16)] * f0;
  float lh  = rhigh[hb + ((size_t)(64 + c) << 16)] * f1;
  float hh_ = rhigh[hb + ((size_t)(128 + c) << 16)] * f2;
  float x1 = (llv - hl - lh + hh_) * 0.5f;
  float x2 = (llv - hl + lh - hh_) * 0.5f;
  float x3 = (llv + hl - lh - hh_) * 0.5f;
  float x4 = (llv + hl + lh + hh_) * 0.5f;
  size_t xb = (((size_t)(b * 64 + c)) * 512 + 2 * h) * 512 + 2 * w;
  df[p][0][c] = x1 - x[xb];
  df[p][1][c] = x3 - x[xb + 1];
  df[p][2][c] = x2 - x[xb + 512];
  df[p][3][c] = x4 - x[xb + 513];
  __syncthreads();
  float a0 = 0.f, a1 = 0.f, a2 = 0.f, a3 = 0.f;
  for (int c2 = 0; c2 < 64; ++c2) {
    float wv = po[c * 64 + c2];
    a0 += wv * df[p][0][c2];
    a1 += wv * df[p][1][c2];
    a2 += wv * df[p][2][c2];
    a3 += wv * df[p][3][c2];
  }
  out[xb] = a0; out[xb + 1] = a1; out[xb + 512] = a2; out[xb + 513] = a3;
}

// ---------------------------------------------------------------------------
extern "C" void kernel_launch(void* const* d_in, const int* in_sizes, int n_in,
                              void* d_out, int out_size, void* d_ws, size_t ws_size,
                              hipStream_t stream) {
  const float* x       = (const float*)d_in[0];
  const float* norm_w  = (const float*)d_in[1];
  const float* norm_b  = (const float*)d_in[2];
  const float* qkv_w   = (const float*)d_in[3];
  const float* qkv_b   = (const float*)d_in[4];
  const float* proj_w  = (const float*)d_in[5];
  const float* proj_b  = (const float*)d_in[6];
  const float* rpb     = (const float*)d_in[7];
  const float* aux_dw  = (const float*)d_in[8];
  const float* aux_pw  = (const float*)d_in[9];
  const float* cg1     = (const float*)d_in[10];
  const float* cg2     = (const float*)d_in[11];
  const float* po_w    = (const float*)d_in[12];
  const float* g_main  = (const float*)d_in[13];
  const float* g_aux   = (const float*)d_in[14];
  const float* g_cross = (const float*)d_in[15];
  const int*   rpi     = (const int*)d_in[16];

  char* ws = (char*)d_ws;
  float*     ll    = (float*)(ws);                     // 8388608 f32
  float*     high  = (float*)(ws + 33554432ull);       // 25165824 f32
  float*     rll   = (float*)(ws + 134217728ull);      // 8388608 f32
  _Float16*  qh    = (_Float16*)(ws + 167772160ull);   // 110592 f16
  _Float16*  ph    = (_Float16*)(ws + 167993344ull);   // 36864 f16
  float*     gap   = (float*)(ws + 168067072ull);      // 128 f32
  float*     crs   = (float*)(ws + 168067584ull);      // 384 f32

  dwt_kernel<<<32768, 256, 0, stream>>>(x, ll, high);
  cvt_kernel<<<432, 256, 0, stream>>>(qkv_w, proj_w, qh, ph);
  gap_kernel<<<128, 256, 0, stream>>>(ll, gap);
  cross_kernel<<<2, 192, 0, stream>>>(gap, cg1, cg2, crs);
  aux_kernel<<<32768, 256, 0, stream>>>(ll, aux_dw, aux_pw, g_aux, rll);
  attn_kernel<<<2048, 128, 0, stream>>>(high, norm_w, norm_b, qh, qkv_b, ph,
                                        proj_b, rpb, rpi, g_main);
  iwt_kernel<<<32768, 256, 0, stream>>>(rll, high, crs, x, po_w, g_cross,
                                        (float*)d_out);
}
